// TernaryLinear_44702019617522
// MI455X (gfx1250) — matrix operand order repaired and checked by reading, not compiled, not run on hardware
//
#include <hip/hip_runtime.h>
#include <stdint.h>

typedef __attribute__((ext_vector_type(16))) _Float16 v16h;
typedef __attribute__((ext_vector_type(8)))  float    v8f;
typedef int vi4 __attribute__((vector_size(16)));     // b128 payload type

#define KDIM   1024   // D_IN
#define NDIM   1024   // D_OUT
#define MTOT   32768  // B*S = 8*4096
#define ALPHA  0.7f

#if defined(__has_builtin)
#if __has_builtin(__builtin_amdgcn_global_load_async_to_lds_b128)
#define USE_ASYNC_LDS 1
#endif
#endif

// ---------------------------------------------------------------------------
// Pass 1: per-block partial sums of |W|  (1024 blocks x 256 threads, 4 elts/thr)
// ---------------------------------------------------------------------------
__global__ __launch_bounds__(256) void absmean_partial(const float* __restrict__ w,
                                                       float* __restrict__ part) {
    __shared__ float sm[256];
    const int i4 = blockIdx.x * 256 + threadIdx.x;          // 0 .. 262143
    float4 v = ((const float4*)w)[i4];
    sm[threadIdx.x] = fabsf(v.x) + fabsf(v.y) + fabsf(v.z) + fabsf(v.w);
    __syncthreads();
    for (int s = 128; s > 0; s >>= 1) {
        if (threadIdx.x < s) sm[threadIdx.x] += sm[threadIdx.x + s];
        __syncthreads();
    }
    if (threadIdx.x == 0) part[blockIdx.x] = sm[0];
}

// Pass 2: reduce 1024 partials -> threshold = ALPHA * mean(|W|)
__global__ __launch_bounds__(256) void absmean_final(const float* __restrict__ part,
                                                     float* __restrict__ thr) {
    __shared__ float sm[256];
    const int t = threadIdx.x;
    sm[t] = part[t] + part[t + 256] + part[t + 512] + part[t + 768];
    __syncthreads();
    for (int s = 128; s > 0; s >>= 1) {
        if (t < s) sm[t] += sm[t + s];
        __syncthreads();
    }
    if (t == 0) thr[0] = ALPHA * sm[0] / (float)(KDIM * NDIM);
}

// ---------------------------------------------------------------------------
// Quantize W (fp32 [N,K]) -> ternary {-1,0,+1} as f16 [N,K] (exact in f16)
// ---------------------------------------------------------------------------
__global__ __launch_bounds__(256) void quantize_w(const float* __restrict__ w,
                                                  const float* __restrict__ thr,
                                                  _Float16* __restrict__ wq) {
    const float th = *thr;
    const int i4 = blockIdx.x * 256 + threadIdx.x;          // 0 .. 262143
    float4 v = ((const float4*)w)[i4];
    const float f[4] = {v.x, v.y, v.z, v.w};
    union { _Float16 h[4]; uint2 u; } pk;
#pragma unroll
    for (int c = 0; c < 4; ++c) {
        float q = (f[c] > th) ? 1.0f : ((f[c] < -th) ? -1.0f : 0.0f);
        pk.h[c] = (_Float16)q;
    }
    ((uint2*)wq)[i4] = pk.u;
}

// ---------------------------------------------------------------------------
// WMMA GEMM: out[m,n] = scale * sum_k x[m,k]*t[n,k] + bias[n]
//   x split into (hi,lo) f16 planes (products vs ternary are exact)
//   block tile 128(M) x 128(N), K-step 32, 256 thr = 8 waves (4M x 2N)
//   All fragment loads are contiguous 16B ds_load_b128 (no half shuffles).
//   B tiles staged via global_load_async_to_lds_b128 (ASYNCcnt).
// ---------------------------------------------------------------------------
__global__ __launch_bounds__(256) void ternary_wmma_gemm(
    const float* __restrict__ x, const _Float16* __restrict__ wq,
    const float* __restrict__ scale, const float* __restrict__ bias,
    float* __restrict__ out) {

    constexpr int SH = 40;                // halfs per LDS row: 80B, 16B-aligned
    __shared__ _Float16 lAhi[2][128 * SH];
    __shared__ _Float16 lAlo[2][128 * SH];
    __shared__ _Float16 lB  [2][128 * SH];

    const int tid  = threadIdx.x;
    const int lane = tid & 31;
    const int wid  = tid >> 5;
    const int wm   = wid & 3;             // 4 wave-slots in M (32 rows each)
    const int wn   = wid >> 2;            // 2 wave-slots in N (64 cols each)
    const int Mbase = blockIdx.y * 128;
    const int Nbase = blockIdx.x * 128;

    // global->LDS staging assignments
    const int ar = tid >> 3, ak = (tid & 7) * 4;   // A: rows ar+32i, float4 of k
    const int br = tid >> 2, bk = (tid & 3) * 8;   // B: rows br+64i, 8 halfs of k

    const float*    xA = x  + (size_t)(Mbase + ar) * KDIM + ak;
    const _Float16* wB = wq + (size_t)(Nbase + br) * KDIM + bk;

    float4 aR[4];

    auto load_A = [&](int kt) {
#pragma unroll
        for (int i = 0; i < 4; ++i)
            aR[i] = *(const float4*)(xA + (size_t)(32 * i) * KDIM + kt * 32);
    };

    auto stage_B = [&](int pp, int kt) {
#ifdef USE_ASYNC_LDS
#pragma unroll
        for (int i = 0; i < 2; ++i) {
            __builtin_amdgcn_global_load_async_to_lds_b128(
                (__attribute__((address_space(1))) vi4*)(void*)(
                    const_cast<_Float16*>(wB) + (size_t)(64 * i) * KDIM + kt * 32),
                (__attribute__((address_space(3))) vi4*)(void*)(
                    &lB[pp][(br + 64 * i) * SH + bk]),
                0, 0);
        }
#else
#pragma unroll
        for (int i = 0; i < 2; ++i) {
            uint4 v = *(const uint4*)(wB + (size_t)(64 * i) * KDIM + kt * 32);
            *(uint4*)&lB[pp][(br + 64 * i) * SH + bk] = v;
        }
#endif
    };

    auto wait_async = [&]() {
#ifdef USE_ASYNC_LDS
#if __has_builtin(__builtin_amdgcn_s_wait_asynccnt)
        __builtin_amdgcn_s_wait_asynccnt(0);
#else
        asm volatile("s_wait_asynccnt 0" ::: "memory");
#endif
#endif
    };

    auto store_A = [&](int pp) {
#pragma unroll
        for (int i = 0; i < 4; ++i) {
            const float f[4] = {aR[i].x, aR[i].y, aR[i].z, aR[i].w};
            union { _Float16 h[4]; uint2 u; } hi, lo;
#pragma unroll
            for (int c = 0; c < 4; ++c) {
                _Float16 h = (_Float16)f[c];
                hi.h[c] = h;
                lo.h[c] = (_Float16)(f[c] - (float)h);
            }
            *(uint2*)&lAhi[pp][(ar + 32 * i) * SH + ak] = hi.u;
            *(uint2*)&lAlo[pp][(ar + 32 * i) * SH + ak] = lo.u;
        }
    };

    v8f acc[2][4];
#pragma unroll
    for (int mt = 0; mt < 2; ++mt)
#pragma unroll
        for (int nt = 0; nt < 4; ++nt) acc[mt][nt] = (v8f){};

    // prologue: stage k-step 0 into buffer 0
    stage_B(0, 0);
    load_A(0);
    store_A(0);
    wait_async();
    __syncthreads();

    const int r  = lane & 15;   // row/col within 16x16 tile
    const int hh = lane >> 4;   // lane half selects K sub-range

    for (int kt = 0; kt < 32; ++kt) {
        const int p = kt & 1;
        if (kt + 1 < 32) {
            stage_B(1 - p, kt + 1);        // async DMA into next buffer
            load_A(kt + 1);                // fp32 x into regs
        }
        if (kt + 2 < 32)
            __builtin_prefetch(xA + (kt + 2) * 32, 0, 1);

        // ---- fragments: each is two contiguous 16B LDS reads ----
        v16h aHi[2], aLo[2], bF[4];
#pragma unroll
        for (int mt = 0; mt < 2; ++mt) {
            const size_t ro = (size_t)(wm * 32 + mt * 16 + r) * (SH * 2) + hh * 16;
            const char* ph = (const char*)&lAhi[p][0] + ro;
            const char* pl = (const char*)&lAlo[p][0] + ro;
            union { uint4 q[2]; v16h h; } fh, fl;
            fh.q[0] = *(const uint4*)(ph);       // k = hh*8 .. hh*8+7
            fh.q[1] = *(const uint4*)(ph + 32);  // k = 16+hh*8 .. +7
            fl.q[0] = *(const uint4*)(pl);
            fl.q[1] = *(const uint4*)(pl + 32);
            aHi[mt] = fh.h;
            aLo[mt] = fl.h;
        }
#pragma unroll
        for (int nt = 0; nt < 4; ++nt) {
            const char* pb = (const char*)&lB[p][0] +
                             (size_t)(wn * 64 + nt * 16 + r) * (SH * 2) + hh * 16;
            union { uint4 q[2]; v16h h; } fb;
            fb.q[0] = *(const uint4*)(pb);       // k = hh*8 .. hh*8+7
            fb.q[1] = *(const uint4*)(pb + 32);  // k = 16+hh*8 .. +7
            bF[nt] = fb.h;
        }

        // ---- 16 WMMAs: acc += a_hi*b + a_lo*b (b in {-1,0,1} -> exact) ----
#pragma unroll
        for (int mt = 0; mt < 2; ++mt)
#pragma unroll
            for (int nt = 0; nt < 4; ++nt) {
                acc[mt][nt] = __builtin_amdgcn_wmma_f32_16x16x32_f16(
                    false, aHi[mt], false, bF[nt], (short)0, acc[mt][nt], false, false);
                acc[mt][nt] = __builtin_amdgcn_wmma_f32_16x16x32_f16(
                    false, aLo[mt], false, bF[nt], (short)0, acc[mt][nt], false, false);
            }

        if (kt + 1 < 32) {
            store_A(1 - p);                // convert + ds_store_b64 x8
            wait_async();                  // B DMA for next buffer complete
        }
        __syncthreads();
    }

    // ---- epilogue: D layout lanes0-15 M=v, lanes16-31 M=8+v; N = base+lane%16
    const float s = scale[0];
#pragma unroll
    for (int mt = 0; mt < 2; ++mt)
#pragma unroll
        for (int nt = 0; nt < 4; ++nt) {
            const int n = Nbase + wn * 64 + nt * 16 + r;
            const float bb = bias[n];
#pragma unroll
            for (int v = 0; v < 8; ++v) {
                const int m = Mbase + wm * 32 + mt * 16 + hh * 8 + v;
                out[(size_t)m * NDIM + n] = s * acc[mt][nt][v] + bb;
            }
        }
}

// ---------------------------------------------------------------------------
extern "C" void kernel_launch(void* const* d_in, const int* in_sizes, int n_in,
                              void* d_out, int out_size, void* d_ws, size_t ws_size,
                              hipStream_t stream) {
    const float* x     = (const float*)d_in[0];   // [8,4096,1024] f32
    const float* w     = (const float*)d_in[1];   // [1024,1024]   f32
    const float* scale = (const float*)d_in[2];   // [1]           f32
    const float* bias  = (const float*)d_in[3];   // [1024]        f32
    float* out = (float*)d_out;                   // [8,4096,1024] f32

    // workspace: [0,4KB) partials, [4KB] threshold, [8KB, 8KB+2MB) wq f16
    float*    part = (float*)d_ws;
    float*    thr  = part + 1024;
    _Float16* wq   = (_Float16*)((char*)d_ws + 8192);

    absmean_partial<<<1024, 256, 0, stream>>>(w, part);
    absmean_final<<<1, 256, 0, stream>>>(part, thr);
    quantize_w<<<1024, 256, 0, stream>>>(w, thr, wq);

    dim3 grid(NDIM / 128, MTOT / 128);   // (8, 256)
    ternary_wmma_gemm<<<grid, 256, 0, stream>>>(x, wq, scale, bias, out);
}
